// Attention_2963527434911
// MI455X (gfx1250) — matrix operand order repaired and checked by reading, not compiled, not run on hardware
//
#include <hip/hip_runtime.h>
#include <hip/hip_bf16.h>

// ---------------------------------------------------------------------------
// MI455X (gfx1250) fused row-attention:
//   q,k,v = 1x1 convs of x[16,8,512,512]; S=QK^T; softmax; out = A @ V_c.
// Roofline: ~39 GFLOP vs ~270MB HBM traffic -> bf16 WMMA (f32 accum) balances
// compute against the 23.3TB/s memory floor; intermediates (~100MB) are
// L2-resident (192MB L2).  GEMM slab staging runs on the Tensor Data Mover
// (tensor_load_to_lds / TENSORcnt), double-buffered against the WMMA loop.
// The TDM writes LDS invisibly to alias analysis, so: (a) the LDS base is
// escaped into an asm once, (b) a memory-clobber asm after each barrier
// forces the fragment ds_loads to actually reload -- while keeping the
// pointers in addrspace(3) so they stay ds_load_b128 (not flat).
// Workspace layout (100,663,296 B total):
//   q   bf16 [16][512][512]   (reused as attn after phase 2)
//   k   bf16 [16][512][512]
//   vT  bf16 [16][8][512(w)][512(h)]
//   S   f32  [16][512][512]
// ---------------------------------------------------------------------------

typedef __bf16 bf16_t;
typedef bf16_t v16bf __attribute__((ext_vector_type(16)));
typedef float  v8f   __attribute__((ext_vector_type(8)));
typedef unsigned int v4u __attribute__((ext_vector_type(4)));
typedef int          v4i __attribute__((ext_vector_type(4)));
typedef int          v8i __attribute__((ext_vector_type(8)));

#define BB 16
#define CC 8
#define HH 512
#define WW 512

#if defined(__has_builtin)
#  if __has_builtin(__builtin_amdgcn_tensor_load_to_lds)
#    define HAVE_TDM 1
#  endif
#endif
#ifndef HAVE_TDM
#  define HAVE_TDM 0
#endif

// ---------------------------------------------------------------------------
// Phase 1: per-pixel q/k/v (1x1 convs) + V transpose through LDS.
// ---------------------------------------------------------------------------
__global__ __launch_bounds__(256) void qkv_kernel(
    const float* __restrict__ x,
    const float* __restrict__ Wq, const float* __restrict__ bq,
    const float* __restrict__ Wk, const float* __restrict__ bk,
    const float* __restrict__ Wv, const float* __restrict__ bv,
    bf16_t* __restrict__ qo, bf16_t* __restrict__ ko, bf16_t* __restrict__ vT)
{
    __shared__ bf16_t vt[CC][32][34];

    const int b = blockIdx.z;
    const int hBase = blockIdx.y * 32;
    const int wBase = blockIdx.x * 32;
    const int t  = threadIdx.x;
    const int wl = t & 31;
    const int hl0 = t >> 5;

    float wq[CC], wk[CC], wv[CC][CC], bvv[CC];
#pragma unroll
    for (int c = 0; c < CC; ++c) { wq[c] = Wq[c]; wk[c] = Wk[c]; }
#pragma unroll
    for (int o = 0; o < CC; ++o) {
        bvv[o] = bv[o];
#pragma unroll
        for (int c = 0; c < CC; ++c) wv[o][c] = Wv[o * CC + c];
    }
    const float bqv = bq[0], bkv = bk[0];
    const size_t plane = (size_t)HH * WW;

#pragma unroll
    for (int hh = 0; hh < 32; hh += 8) {
        const int hl = hl0 + hh;
        const int h  = hBase + hl;
        const int w  = wBase + wl;
        const size_t base = ((size_t)b * CC * HH + h) * WW + w;

        float xv[CC];
#pragma unroll
        for (int c = 0; c < CC; ++c) xv[c] = x[base + (size_t)c * plane];

        float qv = bqv, kv = bkv;
#pragma unroll
        for (int c = 0; c < CC; ++c) { qv += xv[c] * wq[c]; kv += xv[c] * wk[c]; }

        const size_t qkIdx = ((size_t)b * HH + h) * WW + w;
        qo[qkIdx] = (bf16_t)qv;
        ko[qkIdx] = (bf16_t)kv;

#pragma unroll
        for (int o = 0; o < CC; ++o) {
            float vv = bvv[o];
#pragma unroll
            for (int c = 0; c < CC; ++c) vv += xv[c] * wv[o][c];
            vt[o][hl][wl] = (bf16_t)vv;
        }
    }
    __syncthreads();

    const int hc  = t & 31;
    const int wr0 = t >> 5;
#pragma unroll
    for (int wwi = 0; wwi < 32; wwi += 8) {
        const int wr = wr0 + wwi;
#pragma unroll
        for (int c = 0; c < CC; ++c) {
            vT[(((size_t)(b * CC + c)) * WW + (wBase + wr)) * HH + (hBase + hc)] =
                vt[c][hc][wr];
        }
    }
}

// ---------------------------------------------------------------------------
// TDM: stage one 128(row) x 32(col) bf16 slab (row stride 512 elements in
// global) into LDS with 40-element (80B) padded rows, via tensor_load_to_lds.
// D# per ISA ch.8: group0 = {count=1 | lds_addr | global_addr(57b) | type=2},
// group1 = {data_size=2B, pad_enable, pad_interval=3 (every 64B),
//           pad_amount=3 (16B), tensor_dim0/1=512, tile_dim0=32,
//           tile_dim1=128, tensor_dim0_stride=512}.  2-D -> groups 2/3 zero.
// ---------------------------------------------------------------------------
#if HAVE_TDM
__device__ __forceinline__ void tdm_load_tile(const bf16_t* gsrc, unsigned ldsByteOff)
{
    const unsigned long long ga = (unsigned long long)(size_t)gsrc;
    v4u g0;
    g0.x = 1u;                                   // count=1, user mode, no gather
    g0.y = ldsByteOff;                           // LDS byte address (segment base 0)
    g0.z = (unsigned)ga;                         // global_addr[31:0]
    g0.w = (unsigned)((ga >> 32) & 0x01FFFFFFu)  // global_addr[56:32]
           | 0x80000000u;                        // type=2 ("image") in [127:126]
    v8i g1;
    g1[0] = (int)((1u << 16)      // data_size = 2 bytes
                | (1u << 20)      // pad_enable
                | (3u << 22)      // pad_interval: 16 DWORDs = 64B rows
                | (3u << 25));    // pad_amount:   4 DWORDs = 16B pad
    g1[1] = (int)(512u << 16);    // tensor_dim0[15:0] in bits [63:48]
    g1[2] = (int)(512u << 16);    // tensor_dim1[15:0] in bits [95:80]
    g1[3] = (int)(32u  << 16);    // tile_dim0 in bits [127:112]
    g1[4] = 128;                  // tile_dim1 (tile_dim2 = 0)
    g1[5] = 512;                  // tensor_dim0_stride[31:0]
    g1[6] = 0;
    g1[7] = 0;
    const v4i z4 = {0, 0, 0, 0};
#if __clang_major__ >= 23
    const v8i z8 = {0, 0, 0, 0, 0, 0, 0, 0};
    __builtin_amdgcn_tensor_load_to_lds(g0, g1, z4, z4, z8, 0);
#else
    __builtin_amdgcn_tensor_load_to_lds(g0, g1, z4, z4, 0);
#endif
}
#endif

// ---------------------------------------------------------------------------
// D = A * B^T (both row-major, K contiguous), bf16 in, f32 out, M=N=K=512.
// 128x128 tile/WG, 8 waves as 4(M) x 2(N), each wave 2x4 16x16 accumulators.
// Double-buffered LDS slabs staged by the TDM (wave 0 issues, all waves
// consume after s_wait_tensorcnt + barrier).
// ---------------------------------------------------------------------------
__global__ __launch_bounds__(256) void gemm_bt_kernel(
    const bf16_t* __restrict__ A,   // [z/aDiv][512][512]
    const bf16_t* __restrict__ Bm,  // [z][512][512]
    float* __restrict__ D,          // [z][512][512]
    int aDiv)
{
    const int K = 512, N = 512;
    // Single LDS block (assumed at segment offset 0 for the TDM descriptor):
    // [buf][A|B][128*40] bf16 ; per-buffer stride 20480 B, B at +10240 B.
    __shared__ __attribute__((aligned(16))) bf16_t smem[2 * 2 * 128 * 40];

    const int z = blockIdx.z;
    const bf16_t* Ab = A  + (size_t)(z / aDiv) * K * 512;
    const bf16_t* Bb = Bm + (size_t)z * K * 512;
    float*        Db = D  + (size_t)z * N * 512;

    const int tileM = blockIdx.y * 128;
    const int tileN = blockIdx.x * 128;
    const int t    = threadIdx.x;
    const int lane = t & 31;
    const int wid  = t >> 5;
    const int wm   = wid & 3;   // 4 waves along M (32 rows each)
    const int wn   = wid >> 2;  // 2 waves along N (64 cols each)
    const int r    = lane & 15;
    const int half = lane >> 4;

    v8f acc[2][4];
#pragma unroll
    for (int a = 0; a < 2; ++a)
#pragma unroll
        for (int n = 0; n < 4; ++n)
            acc[a][n] = (v8f){0.f, 0.f, 0.f, 0.f, 0.f, 0.f, 0.f, 0.f};

    union Frag { v16bf v; uint4 u[2]; };

#if HAVE_TDM
    // Escape the LDS base once: after this, any memory-clobber asm may have
    // written the shared array, so its loads can't be folded away -- while
    // the load pointers below stay addrspace(3) (ds_load, not flat).
    asm volatile("" : : "v"((void*)smem));
    // Prologue: stage slab 0 into buffer 0.
    if (t < 32) {
        tdm_load_tile(Ab + (size_t)tileM * K, 0u);
        tdm_load_tile(Bb + (size_t)tileN * K, 10240u);
    }
#endif

    int buf = 0;
    for (int k0 = 0; k0 < K; k0 += 32, buf ^= 1) {
        bf16_t* As = smem + buf * 10240;
        bf16_t* Bs = As + 5120;

#if HAVE_TDM
        if (t < 32) __builtin_amdgcn_s_wait_tensorcnt(0);
        __syncthreads();  // LDS slab `buf` now visible to all waves
        asm volatile("" : : : "memory");  // TDM wrote LDS: force reloads below
        if (k0 + 32 < K && t < 32) {
            const unsigned nb = (unsigned)(buf ^ 1) * 20480u;
            tdm_load_tile(Ab + (size_t)tileM * K + k0 + 32, nb);
            tdm_load_tile(Bb + (size_t)tileN * K + k0 + 32, nb + 10240u);
        }
#else
        // Fallback staging: global_load_b128 -> ds_store_b128.
#pragma unroll
        for (int i = 0; i < 2; ++i) {
            const int id  = t + i * 256;
            const int row = id >> 2;
            const int cc  = (id & 3) * 8;
            const uint4 va = *(const uint4*)(Ab + (size_t)(tileM + row) * K + k0 + cc);
            const uint4 vb = *(const uint4*)(Bb + (size_t)(tileN + row) * K + k0 + cc);
            *(uint4*)(As + row * 40 + cc) = va;
            *(uint4*)(Bs + row * 40 + cc) = vb;
        }
        __syncthreads();
#endif

        // Fragment loads per ISA 7.12.2 (wave32):
        //   A: lane r -> row M; K(e) = (e<8 ? e : e+8) + 8*half
        //   B: lane r -> col N; K(e) = (e<8 ? e : e+8) + 8*half
        Frag fa[2];
#pragma unroll
        for (int a = 0; a < 2; ++a) {
            const int row = wm * 32 + a * 16 + r;
            fa[a].u[0] = *(const uint4*)(As + row * 40 + half * 8);
            fa[a].u[1] = *(const uint4*)(As + row * 40 + half * 8 + 16);
        }
        Frag fb[4];
#pragma unroll
        for (int n = 0; n < 4; ++n) {
            const int row = wn * 64 + n * 16 + r;
            fb[n].u[0] = *(const uint4*)(Bs + row * 40 + half * 8);
            fb[n].u[1] = *(const uint4*)(Bs + row * 40 + half * 8 + 16);
        }

#pragma unroll
        for (int a = 0; a < 2; ++a)
#pragma unroll
            for (int n = 0; n < 4; ++n)
                acc[a][n] = __builtin_amdgcn_wmma_f32_16x16x32_bf16(
                    false, fa[a].v, false, fb[n].v,
                    (short)0, acc[a][n], false, false);

        __syncthreads();  // all reads of `buf` done before it is re-staged
    }

    // D tile: row = base + v + 8*half, col = base + (lane&15).
#pragma unroll
    for (int a = 0; a < 2; ++a)
#pragma unroll
        for (int n = 0; n < 4; ++n)
#pragma unroll
            for (int v = 0; v < 8; ++v) {
                const int row = tileM + wm * 32 + a * 16 + half * 8 + v;
                const int col = tileN + wn * 64 + n * 16 + r;
                Db[(size_t)row * N + col] = acc[a][n][v];
            }
}

// ---------------------------------------------------------------------------
// Row softmax: one 256-thread block per row of S (512 f32), output bf16.
// ---------------------------------------------------------------------------
__global__ __launch_bounds__(256) void softmax_kernel(
    const float* __restrict__ S, bf16_t* __restrict__ Aout)
{
    __shared__ float red[256];
    const size_t row = blockIdx.x;
    const float* s = S + row * 512;
    const int t = threadIdx.x;

    const float a = s[t];
    const float b = s[t + 256];

    red[t] = fmaxf(a, b);
    __syncthreads();
    for (int off = 128; off > 0; off >>= 1) {
        if (t < off) red[t] = fmaxf(red[t], red[t + off]);
        __syncthreads();
    }
    const float mx = red[0];
    __syncthreads();

    const float e0 = __expf(a - mx);
    const float e1 = __expf(b - mx);
    red[t] = e0 + e1;
    __syncthreads();
    for (int off = 128; off > 0; off >>= 1) {
        if (t < off) red[t] += red[t + off];
        __syncthreads();
    }
    const float inv = 1.0f / red[0];

    Aout[row * 512 + t]       = (bf16_t)(e0 * inv);
    Aout[row * 512 + t + 256] = (bf16_t)(e1 * inv);
}

// ---------------------------------------------------------------------------
extern "C" void kernel_launch(void* const* d_in, const int* in_sizes, int n_in,
                              void* d_out, int out_size, void* d_ws, size_t ws_size,
                              hipStream_t stream)
{
    const float* x  = (const float*)d_in[0];
    const float* Wq = (const float*)d_in[1];
    const float* bq = (const float*)d_in[2];
    const float* Wk = (const float*)d_in[3];
    const float* bk = (const float*)d_in[4];
    const float* Wv = (const float*)d_in[5];
    const float* bv = (const float*)d_in[6];
    float* out = (float*)d_out;

    const size_t qkElems = (size_t)BB * HH * WW;
    bf16_t* qb = (bf16_t*)d_ws;
    bf16_t* kb = qb + qkElems;
    bf16_t* vT = kb + qkElems;                          // [B][C][W][H]
    float*  S  = (float*)(vT + (size_t)BB * CC * HH * WW);
    bf16_t* attnb = qb;  // q is dead after phase 2; same size as attn.

    qkv_kernel<<<dim3(WW / 32, HH / 32, BB), 256, 0, stream>>>(
        x, Wq, bq, Wk, bk, Wv, bv, qb, kb, vT);

    gemm_bt_kernel<<<dim3(4, 4, BB), 256, 0, stream>>>(qb, kb, S, 1);

    softmax_kernel<<<dim3(BB * HH), 256, 0, stream>>>(S, attnb);

    gemm_bt_kernel<<<dim3(4, 4, BB * CC), 256, 0, stream>>>(attnb, vT, out, CC);
}
